// VanillaSelfAttention_15015205667215
// MI455X (gfx1250) — matrix operand order repaired and checked by reading, not compiled, not run on hardware
//
#include <hip/hip_runtime.h>

typedef _Float16 v16h __attribute__((ext_vector_type(16)));
typedef _Float16 v8h  __attribute__((ext_vector_type(8)));
typedef _Float16 v4h  __attribute__((ext_vector_type(4)));
typedef float    v8f  __attribute__((ext_vector_type(8)));

constexpr int Cdim = 128, Hh = 4, Dd = 32;
constexpr int Bb = 2, Yy = 200, Xx = 200, Nn = Yy * Xx;   // 40000
constexpr int Mtot = Bb * Nn;                              // 80000 rows, /16 = 5000 tiles
constexpr int RS   = 136;  // padded LDS row stride in halves (17*16B -> conflict-free b128)

__device__ __forceinline__ v16h cat8(v8h lo, v8h hi) {
  return __builtin_shufflevector(lo, hi, 0,1,2,3,4,5,6,7,8,9,10,11,12,13,14,15);
}

__device__ __forceinline__ v8f wmma16(v16h a, v16h b, v8f c) {
  // D = A(16x32 f16) * B(32x16 f16) + C(16x16 f32)
  return __builtin_amdgcn_wmma_f32_16x16x32_f16(false, a, false, b, (short)0, c, false, false);
}

// ---------------------------------------------------------------------------
// Pack a row-major f32 weight (128 x ncols) into f16 WMMA B-fragments:
// frag index = nt*4 + kb  (nt = 16-col tile, kb = 32-deep K step)
// lane L (g = L/16, n0 = L%16) holds halves i=0..15 with K = kb*32 + 16*(i/8) + g*8 + i%8
// (same k order as the A fragment: halves 0..7 K=g*8.., halves 8..15 K=16+g*8..)
// ---------------------------------------------------------------------------
__global__ void pack_weights_kernel(const float* __restrict__ src,
                                    _Float16* __restrict__ dst, int ncols) {
  const int frag = blockIdx.x;            // nt*4 + kb
  const int nt = frag >> 2, kb = frag & 3;
  const int L = threadIdx.x;
  const int g = L >> 4, n0 = L & 15;
  _Float16* d = dst + ((size_t)frag * 32 + L) * 16;
  #pragma unroll
  for (int i = 0; i < 16; ++i) {
    const int K = kb * 32 + (i >> 3) * 16 + g * 8 + (i & 7);
    d[i] = (_Float16)src[(size_t)K * ncols + nt * 16 + n0];
  }
}

// A-fragment builder: lane L (g=L/16, m=L%16), halves 0..7 = row[kb*32+g*8 .. +7],
// halves 8..15 = row[kb*32+16+g*8 .. +7]   (ISA 16-bit A 16x32 layout)
__device__ __forceinline__ v16h load_afrag(const _Float16* at, int m, int g, int kb) {
  v8h lo = *(const v8h*)&at[m * RS + kb * 32 + g * 8];
  v8h hi = *(const v8h*)&at[m * RS + kb * 32 + 16 + g * 8];
  return cat8(lo, hi);
}

// ---------------------------------------------------------------------------
// Kernel 1: q = query + query_pos;  value = q@Wv+bv  -> (B,H,N,D) scatter
//           off = q@Woff+boff -> (B*N,64);  logits = q@Wattn+battn -> (B*N,32)
// One wave per 16-row tile, 8 waves / block.
// ---------------------------------------------------------------------------
__global__ __launch_bounds__(256) void proj_kernel(
    const float* __restrict__ query, const float* __restrict__ qpos,
    const _Float16* __restrict__ wv_pk, const float* __restrict__ bv,
    const _Float16* __restrict__ wf_pk, const float* __restrict__ boff,
    const _Float16* __restrict__ wa_pk, const float* __restrict__ battn,
    float* __restrict__ value, float* __restrict__ offbuf,
    float* __restrict__ logits) {
  __shared__ _Float16 tile[8][16 * RS];
  const int wave = threadIdx.x >> 5, lane = threadIdx.x & 31;
  const int rowBase = (blockIdx.x * 8 + wave) * 16;
  _Float16* at = tile[wave];

  // Stage q+qpos tile (16 x 128) into LDS as f16, coalesced float4 loads.
  const float4* q4 = (const float4*)query;
  const float4* p4 = (const float4*)qpos;
  #pragma unroll
  for (int it = 0; it < 16; ++it) {
    const int slot = it * 32 + lane;   // 0..511 float4 slots
    const int r = slot >> 5, c4 = slot & 31;
    const size_t gi = (size_t)(rowBase + r) * 32 + c4;
    const float4 qa = q4[gi], qb = p4[gi];
    v4h hv;
    hv[0] = (_Float16)(qa.x + qb.x); hv[1] = (_Float16)(qa.y + qb.y);
    hv[2] = (_Float16)(qa.z + qb.z); hv[3] = (_Float16)(qa.w + qb.w);
    *(v4h*)&at[r * RS + c4 * 4] = hv;
  }
  __syncthreads();

  const int g = lane >> 4, m = lane & 15;
  v16h afr[4];
  #pragma unroll
  for (int kb = 0; kb < 4; ++kb) afr[kb] = load_afrag(at, m, g, kb);

  // ---- value projection (8 col-tiles) ----
  #pragma unroll
  for (int nt = 0; nt < 8; ++nt) {
    v8f acc = {};
    #pragma unroll
    for (int kb = 0; kb < 4; ++kb) {
      v16h bf = *(const v16h*)(wv_pk + ((size_t)(nt * 4 + kb) * 32 + lane) * 16);
      acc = wmma16(afr[kb], bf, acc);
    }
    const int col = nt * 16 + m;
    const float bias = bv[col];
    const int hh = col >> 5, dd = col & 31;
    #pragma unroll
    for (int r = 0; r < 8; ++r) {
      const int gr = rowBase + r + 8 * g;          // D-tile row M = r + 8*(lane/16)
      const int bb = gr / Nn, nn = gr - bb * Nn;
      value[(((size_t)bb * Hh + hh) * Nn + nn) * Dd + dd] = acc[r] + bias;
    }
  }
  // ---- offset projection (4 col-tiles -> 64 cols) ----
  #pragma unroll
  for (int nt = 0; nt < 4; ++nt) {
    v8f acc = {};
    #pragma unroll
    for (int kb = 0; kb < 4; ++kb) {
      v16h bf = *(const v16h*)(wf_pk + ((size_t)(nt * 4 + kb) * 32 + lane) * 16);
      acc = wmma16(afr[kb], bf, acc);
    }
    const int col = nt * 16 + m;
    const float bias = boff[col];
    #pragma unroll
    for (int r = 0; r < 8; ++r) {
      const int gr = rowBase + r + 8 * g;
      offbuf[(size_t)gr * 64 + col] = acc[r] + bias;
    }
  }
  // ---- attention logits (2 col-tiles -> 32 cols) ----
  #pragma unroll
  for (int nt = 0; nt < 2; ++nt) {
    v8f acc = {};
    #pragma unroll
    for (int kb = 0; kb < 4; ++kb) {
      v16h bf = *(const v16h*)(wa_pk + ((size_t)(nt * 4 + kb) * 32 + lane) * 16);
      acc = wmma16(afr[kb], bf, acc);
    }
    const int col = nt * 16 + m;
    const float bias = battn[col];
    #pragma unroll
    for (int r = 0; r < 8; ++r) {
      const int gr = rowBase + r + 8 * g;
      logits[(size_t)gr * 32 + col] = acc[r] + bias;
    }
  }
}

// ---------------------------------------------------------------------------
// Kernel 2: softmax over P + bilinear gather + weighted sum.
// One wave per (b,n,h); lane = channel d. Gathers are 128B coalesced.
// ---------------------------------------------------------------------------
__global__ __launch_bounds__(256) void sample_kernel(
    const float* __restrict__ value, const float* __restrict__ offbuf,
    const float* __restrict__ logits, float* __restrict__ out1) {
  const int wave = threadIdx.x >> 5, lane = threadIdx.x & 31;
  const int idx = blockIdx.x * 8 + wave;     // flat (bn, h)
  const int h = idx & (Hh - 1);
  const int bn = idx >> 2;
  const int b = bn / Nn, n = bn - b * Nn;
  const int y = n / Xx, x = n - y * Xx;
  const float ry = (y + 0.5f) / Yy;          // ref[...,0] = ry (faithful quirk)
  const float rx = (x + 0.5f) / Xx;          // ref[...,1] = rx

  float l[8];
  const float* lg = logits + (size_t)bn * 32 + h * 8;
  float mx = -3.0e38f;
  #pragma unroll
  for (int p = 0; p < 8; ++p) { l[p] = lg[p]; mx = fmaxf(mx, l[p]); }
  float s = 0.f;
  #pragma unroll
  for (int p = 0; p < 8; ++p) { l[p] = __expf(l[p] - mx); s += l[p]; }
  const float inv = 1.0f / s;

  const float* vb = value + ((size_t)b * Hh + h) * (size_t)Nn * Dd + lane;
  const float* ofs = offbuf + (size_t)bn * 64 + h * 16;
  float acc = 0.f;
  #pragma unroll
  for (int p = 0; p < 8; ++p) {
    const float o0 = ofs[p * 2 + 0], o1 = ofs[p * 2 + 1];
    // loc0 = ry + off0/X; gx = loc0*X - 0.5   (channel 0 sampled as x)
    const float gx = (ry + o0 * (1.0f / Xx)) * (float)Xx - 0.5f;
    const float gy = (rx + o1 * (1.0f / Yy)) * (float)Yy - 0.5f;
    const float x0f = floorf(gx), y0f = floorf(gy);
    const float wx1 = gx - x0f, wy1 = gy - y0f;
    const int x0 = (int)x0f, y0 = (int)y0f;
    const float wp = l[p] * inv;
    float r = 0.f;
    #pragma unroll
    for (int cy = 0; cy < 2; ++cy) {
      #pragma unroll
      for (int cx = 0; cx < 2; ++cx) {
        const int yi = y0 + cy, xi = x0 + cx;
        const float w = (cy ? wy1 : 1.f - wy1) * (cx ? wx1 : 1.f - wx1);
        const bool valid = (yi >= 0) && (yi < Yy) && (xi >= 0) && (xi < Xx);
        const int yc = min(max(yi, 0), Yy - 1), xc = min(max(xi, 0), Xx - 1);
        const float v = vb[(size_t)(yc * Xx + xc) * Dd];
        r += (valid ? w : 0.f) * v;
      }
    }
    acc += wp * r;
  }
  out1[((size_t)b * Nn + n) * Cdim + h * Dd + lane] = acc;
}

// ---------------------------------------------------------------------------
// Kernel 3: out = (out1@Wout1+b1)@Wout2+b2 + residual. Two chained WMMA GEMMs
// through the same per-wave LDS tile.
// ---------------------------------------------------------------------------
__global__ __launch_bounds__(256) void out_kernel(
    const float* __restrict__ out1,
    const _Float16* __restrict__ w1_pk, const float* __restrict__ b1,
    const _Float16* __restrict__ w2_pk, const float* __restrict__ b2,
    const float* __restrict__ residual, float* __restrict__ outp) {
  __shared__ _Float16 tile[8][16 * RS];
  const int wave = threadIdx.x >> 5, lane = threadIdx.x & 31;
  const int rowBase = (blockIdx.x * 8 + wave) * 16;
  _Float16* at = tile[wave];

  const float4* s4 = (const float4*)out1;
  #pragma unroll
  for (int it = 0; it < 16; ++it) {
    const int slot = it * 32 + lane;
    const int r = slot >> 5, c4 = slot & 31;
    const float4 a = s4[(size_t)(rowBase + r) * 32 + c4];
    v4h hv;
    hv[0] = (_Float16)a.x; hv[1] = (_Float16)a.y;
    hv[2] = (_Float16)a.z; hv[3] = (_Float16)a.w;
    *(v4h*)&at[r * RS + c4 * 4] = hv;
  }
  __syncthreads();

  const int g = lane >> 4, m = lane & 15;
  v16h afr[4];
  #pragma unroll
  for (int kb = 0; kb < 4; ++kb) afr[kb] = load_afrag(at, m, g, kb);
  __syncthreads();

  // GEMM1: tmp = out1@Wout1 + b1  -> overwrite LDS tile as f16
  #pragma unroll
  for (int nt = 0; nt < 8; ++nt) {
    v8f acc = {};
    #pragma unroll
    for (int kb = 0; kb < 4; ++kb) {
      v16h bf = *(const v16h*)(w1_pk + ((size_t)(nt * 4 + kb) * 32 + lane) * 16);
      acc = wmma16(afr[kb], bf, acc);
    }
    const int col = nt * 16 + m;
    const float bias = b1[col];
    #pragma unroll
    for (int r = 0; r < 8; ++r)
      at[(r + 8 * g) * RS + col] = (_Float16)(acc[r] + bias);
  }
  __syncthreads();

  v16h tfr[4];
  #pragma unroll
  for (int kb = 0; kb < 4; ++kb) tfr[kb] = load_afrag(at, m, g, kb);

  // GEMM2: out = tmp@Wout2 + b2 + residual
  #pragma unroll
  for (int nt = 0; nt < 8; ++nt) {
    v8f acc = {};
    #pragma unroll
    for (int kb = 0; kb < 4; ++kb) {
      v16h bf = *(const v16h*)(w2_pk + ((size_t)(nt * 4 + kb) * 32 + lane) * 16);
      acc = wmma16(tfr[kb], bf, acc);
    }
    const int col = nt * 16 + m;
    const float bias = b2[col];
    #pragma unroll
    for (int r = 0; r < 8; ++r) {
      const int gr = rowBase + r + 8 * g;
      const size_t o = (size_t)gr * Cdim + col;
      outp[o] = acc[r] + bias + residual[o];
    }
  }
}

// ---------------------------------------------------------------------------
extern "C" void kernel_launch(void* const* d_in, const int* in_sizes, int n_in,
                              void* d_out, int out_size, void* d_ws, size_t ws_size,
                              hipStream_t stream) {
  (void)in_sizes; (void)n_in; (void)out_size; (void)ws_size;
  const float* query = (const float*)d_in[0];
  const float* qpos  = (const float*)d_in[1];
  const float* Wv    = (const float*)d_in[2];
  const float* bv    = (const float*)d_in[3];
  const float* Woff  = (const float*)d_in[4];
  const float* boff  = (const float*)d_in[5];
  const float* Wattn = (const float*)d_in[6];
  const float* battn = (const float*)d_in[7];
  const float* Wout1 = (const float*)d_in[8];
  const float* bout1 = (const float*)d_in[9];
  const float* Wout2 = (const float*)d_in[10];
  const float* bout2 = (const float*)d_in[11];
  float* outp = (float*)d_out;

  char* ws = (char*)d_ws;
  _Float16* wv_pk = (_Float16*)ws;           // 8 tiles * 4 k * 32 * 16 = 16384 halves
  _Float16* wf_pk = wv_pk + 8 * 4 * 32 * 16;
  _Float16* wa_pk = wf_pk + 4 * 4 * 32 * 16;
  _Float16* w1_pk = wa_pk + 2 * 4 * 32 * 16;
  _Float16* w2_pk = w1_pk + 8 * 4 * 32 * 16;
  float* value  = (float*)(ws + 131072);                   // B*H*N*D f32
  float* offbuf = value + (size_t)Bb * Hh * Nn * Dd;       // (B*N,64)
  float* logits = offbuf + (size_t)Mtot * 64;              // (B*N,32)
  float* out1   = logits + (size_t)Mtot * 32;              // (B*N,128)

  pack_weights_kernel<<<32, 32, 0, stream>>>(Wv,    wv_pk, 128);
  pack_weights_kernel<<<16, 32, 0, stream>>>(Woff,  wf_pk, 64);
  pack_weights_kernel<<< 8, 32, 0, stream>>>(Wattn, wa_pk, 32);
  pack_weights_kernel<<<32, 32, 0, stream>>>(Wout1, w1_pk, 128);
  pack_weights_kernel<<<32, 32, 0, stream>>>(Wout2, w2_pk, 128);

  proj_kernel<<<Mtot / 128, 256, 0, stream>>>(query, qpos, wv_pk, bv, wf_pk, boff,
                                              wa_pk, battn, value, offbuf, logits);
  sample_kernel<<<(Bb * Nn * Hh) / 8, 256, 0, stream>>>(value, offbuf, logits, out1);
  out_kernel<<<Mtot / 128, 256, 0, stream>>>(out1, w1_pk, bout1, w2_pk, bout2,
                                             query, outp);
}
